// DGCNNModule_54872502174370
// MI455X (gfx1250) — matrix operand order repaired and checked by reading, not compiled, not run on hardware
//
#include <hip/hip_runtime.h>
#include <hip/hip_bf16.h>
#include <stdint.h>

// ---------------------------------------------------------------------------
// DGCNN pipeline for MI455X (gfx1250, wave32, WMMA + TDM).
// All GEMM-shaped work (Gram matrices for kNN + all MLP layers) runs through
// v_wmma_f32_16x16x32_f16.  MLP weights are staged into LDS per-workgroup via
// the Tensor Data Mover (tensor_load_to_lds / s_wait_tensorcnt).  BN stats are
// fused into GEMM epilogues; BN affine folded into following layers' weights.
// ---------------------------------------------------------------------------

typedef _Float16 v16h __attribute__((ext_vector_type(16)));
typedef _Float16 v2h  __attribute__((ext_vector_type(2)));
typedef float    v8f  __attribute__((ext_vector_type(8)));
typedef float    v4f  __attribute__((ext_vector_type(4)));
typedef unsigned int u32x4 __attribute__((ext_vector_type(4)));
typedef int          i32x8 __attribute__((ext_vector_type(8)));
typedef int          i32x4 __attribute__((ext_vector_type(4)));

union AH  { uint32_t u[8]; v16h v; };   // A-matrix fragment (16 f16 / lane)
union BH  { v4f f[2];      v16h v; };   // B-matrix fragment (16 f16 / lane)
union UH2 { uint32_t u;    v2h  h; };   // packed f16 pair

// K offset inside a 32-wide chunk for A-fragment element pair v (lane half h):
// lanes 0-15: VGPR0-7 -> K = 0,2,...,14 ; lanes16-31: VGPR0-7 -> K = 16,...,30
__device__ __forceinline__ int a_koff(int v, int half) {
    return 2 * v + 16 * half;
}

// ---------------------------------------------------------------------------
// Tensor Data Mover: DMA a dense 2-D block (rows x dwordsPerRow dwords,
// contiguous) from global memory into LDS.  D# layout per CDNA5 ISA §8.3/8.4:
//   group0: count=1 | lds_addr | global_addr[56:0] | type=2
//   group1: data_size=2 (4B) | tensor_dim0/1 | tile_dim0/1 | dim0 stride
// This toolchain's builtin takes all four D# groups (g2/g3 + trailing i32x8
// group) plus a cache-policy immediate: 6 arguments total.
__device__ __forceinline__ void tdm_load_2d_to_lds(const void* gptr, uint32_t lds_off,
                                                   int dwordsPerRow, int rows) {
    uint64_t ga = (uint64_t)(uintptr_t)gptr;
    u32x4 g0;
    g0[0] = 1u;                                              // count=1 (valid user D#)
    g0[1] = lds_off;                                         // LDS byte address
    g0[2] = (uint32_t)ga;                                    // global_addr[31:0]
    g0[3] = (uint32_t)((ga >> 32) & 0x01FFFFFFu) | (2u << 30); // global_addr[56:32], type=2
    i32x8 g1;
    g1[0] = (int)(2u << 16);                                 // data_size=2 -> 4-byte elems
    g1[1] = (int)((unsigned)(dwordsPerRow & 0xFFFF) << 16);  // tensor_dim0[15:0]
    g1[2] = (int)(((unsigned)(dwordsPerRow >> 16) & 0xFFFFu) |
                  ((unsigned)(rows & 0xFFFF) << 16));        // tensor_dim0[31:16] | tensor_dim1[15:0]
    g1[3] = (int)(((unsigned)(rows >> 16) & 0xFFFFu) |
                  ((unsigned)(dwordsPerRow & 0xFFFF) << 16)); // tensor_dim1[31:16] | tile_dim0
    g1[4] = (int)(rows & 0xFFFF);                            // tile_dim1 (tile_dim2=0)
    g1[5] = dwordsPerRow;                                    // tensor_dim0_stride[31:0]
    g1[6] = 0;                                               // stride hi | dim1_stride lo
    g1[7] = 0;
    i32x4 gz4 = {0, 0, 0, 0};
    i32x8 gz8 = {0, 0, 0, 0, 0, 0, 0, 0};
    __builtin_amdgcn_tensor_load_to_lds(g0, g1, gz4, gz4, gz8, 0);
}

// ---------------------------------------------------------------------------
// Convert f32 features -> f16, compute squared row norms.
__global__ void cvt_sq_kernel(const float* __restrict__ X, _Float16* __restrict__ Xh,
                              float* __restrict__ sqn, int rows, int cols) {
    int r = blockIdx.x * blockDim.x + threadIdx.x;
    if (r >= rows) return;
    const float* src = X + (size_t)r * cols;
    _Float16* dst = Xh + (size_t)r * cols;
    float s = 0.f;
    for (int c = 0; c < cols; ++c) { float v = src[c]; s += v * v; dst[c] = (_Float16)v; }
    sqn[r] = s;
}

// Squared row norms of a strided f16 matrix (x1 block inside hcat).
__global__ void sqnorm_h_kernel(const _Float16* __restrict__ Xh, int stride, int cols,
                                float* __restrict__ sqn, int rows) {
    int r = blockIdx.x * blockDim.x + threadIdx.x;
    if (r >= rows) return;
    const _Float16* p = Xh + (size_t)r * stride;
    float s = 0.f;
    for (int c = 0; c < cols; ++c) { float v = (float)p[c]; s += v * v; }
    sqn[r] = s;
}

// ---------------------------------------------------------------------------
// Pairwise squared distances per batch via WMMA Gram tiles.
// D[b*N+i][j] = |x_i|^2 + |x_j|^2 - 2 <x_i,x_j>.   N = 1024 fixed.
__global__ void gram_dist_kernel(const _Float16* __restrict__ X, int stride, int C,
                                 const float* __restrict__ sqn,
                                 float* __restrict__ D, int tilesPerDim) {
    int wave = (blockIdx.x * blockDim.x + threadIdx.x) >> 5;
    int lane = threadIdx.x & 31;
    int half = lane >> 4, l16 = lane & 15;
    int tj = wave % tilesPerDim;
    int ti = (wave / tilesPerDim) % tilesPerDim;
    int b  = wave / (tilesPerDim * tilesPerDim);
    int rowBase = b << 10;
    const _Float16* rowA = X + (size_t)(rowBase + ti * 16 + l16) * stride;
    const _Float16* rowB = X + (size_t)(rowBase + tj * 16 + l16) * stride;
    v8f acc = {0.f, 0.f, 0.f, 0.f, 0.f, 0.f, 0.f, 0.f};
    for (int k0 = 0; k0 < C; k0 += 32) {
        AH a;
#pragma unroll
        for (int v = 0; v < 8; ++v)
            a.u[v] = *reinterpret_cast<const uint32_t*>(rowA + k0 + a_koff(v, half));
        BH bm;
        const v4f* bp = reinterpret_cast<const v4f*>(rowB + k0 + 16 * half);
        bm.f[0] = bp[0]; bm.f[1] = bp[1];
        acc = __builtin_amdgcn_wmma_f32_16x16x32_f16(false, a.v, false, bm.v,
                                                     (short)0, acc, false, false);
    }
    int j = tj * 16 + l16;
    float sqj = sqn[rowBase + j];
#pragma unroll
    for (int vr = 0; vr < 8; ++vr) {
        int i = ti * 16 + vr + 8 * half;
        float d = sqn[rowBase + i] + sqj - 2.0f * acc[vr];
        D[((size_t)(rowBase + i) << 10) + j] = d;
    }
}

// ---------------------------------------------------------------------------
// Per-row top-20 smallest distances (self included, as in reference).
__global__ void topk_kernel(const float* __restrict__ D, int* __restrict__ idx) {
    int r = blockIdx.x * blockDim.x + threadIdx.x;
    if (r >= 8192) return;
    const float* row = D + ((size_t)r << 10);
    float bd[20]; int bi[20];
#pragma unroll
    for (int t = 0; t < 20; ++t) { bd[t] = 3.4e38f; bi[t] = 0; }
    for (int j = 0; j < 1024; ++j) {
        float d = row[j];
        if (d < bd[19]) {
            int t = 19;
            while (t > 0 && bd[t - 1] > d) { bd[t] = bd[t - 1]; bi[t] = bi[t - 1]; --t; }
            bd[t] = d; bi[t] = j;
        }
    }
    for (int t = 0; t < 20; ++t) idx[r * 20 + t] = bi[t];
}

// ---------------------------------------------------------------------------
// Generic WMMA GEMM:  Out = relu(A @ W + b), f16 out, fused BN-stat atomics.
// Weights (Wt, f16 [Nout][K]) are staged into LDS once per workgroup via TDM;
// every wave then reads its B-fragments from LDS (ds_load_b128).
// EDGE==1: A rows are edge features [x_i | x_j - x_i] built on the fly from
//          `A` (feature rows, stride featStride, Cfeat cols) and `idx`.
// EDGE==0: A is a dense [M][K] f16 matrix.
template <int EDGE>
__global__ void gemm_kernel(const _Float16* __restrict__ A, const int* __restrict__ idx,
                            int featStride, int Cfeat,
                            const _Float16* __restrict__ Wt, const float* __restrict__ bias,
                            _Float16* __restrict__ Out,
                            float* __restrict__ gsum, float* __restrict__ gsumsq,
                            int K, int Nout, int nTilesN, int totalTiles) {
    extern __shared__ _Float16 smem[];   // Nout*K f16 weight block

    // Stage the whole weight matrix into LDS with the Tensor Data Mover.
    // Wave 0 issues the DMA and waits on TENSORcnt; barrier releases the rest.
    if (threadIdx.x < 32) {
        tdm_load_2d_to_lds(Wt, (uint32_t)(uintptr_t)smem, K >> 1, Nout);
        __builtin_amdgcn_s_wait_tensorcnt(0);
    }
    __syncthreads();

    int wave = (blockIdx.x * blockDim.x + threadIdx.x) >> 5;
    if (wave >= totalTiles) return;          // wave-uniform guard (EXEC stays all-1)
    int lane = threadIdx.x & 31;
    int half = lane >> 4, l16 = lane & 15;
    int tn = wave % nTilesN, tm = wave / nTilesN;
    int m = tm * 16 + l16;

    const _Float16* rowI;
    const _Float16* rowJ = nullptr;
    if (EDGE) {
        int r  = m / 20;                     // global point row (b*1024 + i)
        int j  = idx[m];                     // within-batch neighbor
        int jg = (r & ~1023) + j;            // global neighbor row
        rowI = A + (size_t)r  * featStride;
        rowJ = A + (size_t)jg * featStride;
    } else {
        rowI = A + (size_t)m * K;
    }
    const _Float16* wrow = smem + (size_t)(tn * 16 + l16) * K;

    v8f acc = {0.f, 0.f, 0.f, 0.f, 0.f, 0.f, 0.f, 0.f};
    for (int k0 = 0; k0 < K; k0 += 32) {
        AH a;
        if (EDGE) {
            if (k0 < Cfeat) {                // first half: x_i
#pragma unroll
                for (int v = 0; v < 8; ++v)
                    a.u[v] = *reinterpret_cast<const uint32_t*>(rowI + k0 + a_koff(v, half));
            } else {                         // second half: x_j - x_i (packed f16)
                int base = k0 - Cfeat;
#pragma unroll
                for (int v = 0; v < 8; ++v) {
                    int kb = a_koff(v, half);
                    UH2 pi, pj, d;
                    pi.u = *reinterpret_cast<const uint32_t*>(rowI + base + kb);
                    pj.u = *reinterpret_cast<const uint32_t*>(rowJ + base + kb);
                    d.h = pj.h - pi.h;
                    a.u[v] = d.u;
                }
            }
        } else {
#pragma unroll
            for (int v = 0; v < 8; ++v)
                a.u[v] = *reinterpret_cast<const uint32_t*>(rowI + k0 + a_koff(v, half));
        }
        BH bm;
        const v4f* bp = reinterpret_cast<const v4f*>(wrow + k0 + 16 * half);
        bm.f[0] = bp[0]; bm.f[1] = bp[1];
        acc = __builtin_amdgcn_wmma_f32_16x16x32_f16(false, a.v, false, bm.v,
                                                     (short)0, acc, false, false);
    }

    int n = tn * 16 + l16;
    float bv = bias[n];
    float s = 0.f, sq = 0.f;
#pragma unroll
    for (int vr = 0; vr < 8; ++vr) {
        float x = acc[vr] + bv;
        x = fmaxf(x, 0.f);                   // ReLU
        s += x; sq += x * x;
        int mr = tm * 16 + vr + 8 * half;
        Out[(size_t)mr * Nout + n] = (_Float16)x;
    }
    // BN statistics: reduce lane pairs sharing a column, then 2 atomics.
    s  += __shfl_xor(s, 16, 32);
    sq += __shfl_xor(sq, 16, 32);
    if (half == 0) {
        atomicAdd(&gsum[n], s);
        atomicAdd(&gsumsq[n], sq);
    }
}

// ---------------------------------------------------------------------------
// BN finalize: per-channel scale/shift from accumulated sums.
__global__ void bn_finalize(const float* __restrict__ gsum, const float* __restrict__ gsumsq,
                            const float* __restrict__ g, const float* __restrict__ be,
                            float* __restrict__ sArr, float* __restrict__ tArr,
                            int ch, float invM) {
    int c = blockIdx.x * blockDim.x + threadIdx.x;
    if (c >= ch) return;
    float mean = gsum[c] * invM;
    float var  = gsumsq[c] * invM - mean * mean;
    float sc   = g[c] * rsqrtf(var + 1e-5f);
    sArr[c] = sc;
    tArr[c] = be[c] - mean * sc;
}

// Weight prep (no BN fold): Wt[n][k] = (f16) W[k][n]; bEff = b.
__global__ void prep_w_plain(const float* __restrict__ W, const float* __restrict__ b,
                             _Float16* __restrict__ Wt, float* __restrict__ bEff,
                             int K, int Nout) {
    int t = blockIdx.x * blockDim.x + threadIdx.x;
    if (t < K * Nout) {
        int n = t / K, k = t % K;
        Wt[t] = (_Float16)W[(size_t)k * Nout + n];
    }
    if (t < Nout) bEff[t] = b[t];
}

// Weight prep with previous-layer BN folded in: Wt[n][k] = s[k]*W[k][n].
__global__ void prep_w_folded(const float* __restrict__ W, _Float16* __restrict__ Wt,
                              const float* __restrict__ sArr, int K, int Nout) {
    int t = blockIdx.x * blockDim.x + threadIdx.x;
    if (t >= K * Nout) return;
    int n = t / K, k = t % K;
    Wt[t] = (_Float16)(sArr[k] * W[(size_t)k * Nout + n]);
}
__global__ void prep_b_folded(const float* __restrict__ W, const float* __restrict__ b,
                              const float* __restrict__ tArr, float* __restrict__ bEff,
                              int K, int Nout) {
    int n = blockIdx.x * blockDim.x + threadIdx.x;
    if (n >= Nout) return;
    float acc = b[n];
    for (int k = 0; k < K; ++k) acc += tArr[k] * W[(size_t)k * Nout + n];
    bEff[n] = acc;
}

// Max over k neighbors, then BN affine, write into hcat (stride 192).
__global__ void aggregate_kernel(const _Float16* __restrict__ H, int ch,
                                 const float* __restrict__ sArr, const float* __restrict__ tArr,
                                 _Float16* __restrict__ hcat, int colOff, int rows) {
    int t = blockIdx.x * blockDim.x + threadIdx.x;
    if (t >= rows * ch) return;
    int r = t / ch, c = t % ch;
    const _Float16* p = H + (size_t)(r * 20) * ch + c;
    float m = -3.4e38f;
    for (int kk = 0; kk < 20; ++kk) m = fmaxf(m, (float)p[(size_t)kk * ch]);
    float res = sArr[c] * m + tArr[c];
    hcat[(size_t)r * 192 + colOff + c] = (_Float16)res;
}

// Final BN apply -> f32 output.
__global__ void apply_kernel(const _Float16* __restrict__ hF,
                             const float* __restrict__ sArr, const float* __restrict__ tArr,
                             float* __restrict__ out, int total, int ch) {
    int t = blockIdx.x * blockDim.x + threadIdx.x;
    if (t >= total) return;
    int c = t % ch;
    out[t] = sArr[c] * (float)hF[t] + tArr[c];
}

__global__ void zero_kernel(float* __restrict__ p, int n) {
    int t = blockIdx.x * blockDim.x + threadIdx.x;
    if (t < n) p[t] = 0.f;
}

// ---------------------------------------------------------------------------
extern "C" void kernel_launch(void* const* d_in, const int* in_sizes, int n_in,
                              void* d_out, int out_size, void* d_ws, size_t ws_size,
                              hipStream_t stream) {
    (void)in_sizes; (void)n_in; (void)out_size; (void)ws_size;
    const float* feat   = (const float*)d_in[1];
    const float* c1_w0  = (const float*)d_in[2];
    const float* c1_b0  = (const float*)d_in[3];
    const float* c1_g0  = (const float*)d_in[4];
    const float* c1_be0 = (const float*)d_in[5];
    const float* c1_w1  = (const float*)d_in[6];
    const float* c1_b1  = (const float*)d_in[7];
    const float* c1_g1  = (const float*)d_in[8];
    const float* c1_be1 = (const float*)d_in[9];
    const float* c1_w2  = (const float*)d_in[10];
    const float* c1_b2  = (const float*)d_in[11];
    const float* c1_g2  = (const float*)d_in[12];
    const float* c1_be2 = (const float*)d_in[13];
    const float* c2_w0  = (const float*)d_in[14];
    const float* c2_b0  = (const float*)d_in[15];
    const float* c2_g0  = (const float*)d_in[16];
    const float* c2_be0 = (const float*)d_in[17];
    const float* l_w    = (const float*)d_in[18];
    const float* l_b    = (const float*)d_in[19];
    const float* l_g    = (const float*)d_in[20];
    const float* l_be   = (const float*)d_in[21];
    float* out = (float*)d_out;

    // Workspace layout (all offsets 256B-aligned).
    char* w = (char*)d_ws;
    _Float16* xh    = (_Float16*)(w + 0);          // 8192x256 f16        (4 MB)
    float*    sqn   = (float*)   (w + 4194304);    // 8192 f32
    int*      idx   = (int*)     (w + 4227072);    // 163840 int
    _Float16* hcat  = (_Float16*)(w + 4882432);    // 8192x192 f16        (3 MB)
    _Float16* hF    = (_Float16*)(w + 8028160);    // 8192x128 f16        (2 MB)
    _Float16* Wt    = (_Float16*)(w + 10125312);   // max 128x512 f16
    float*    bEff  = (float*)   (w + 10256384);   // 128 f32
    float*    gsum  = (float*)   (w + 10256896);   // 128 f32 (gsumsq follows)
    float*    gsumsq= (float*)   (w + 10257408);   // 128 f32
    float*    sArr  = (float*)   (w + 10257920);   // 128 f32
    float*    tArr  = (float*)   (w + 10258432);   // 128 f32
    float*    D     = (float*)   (w + 12582912);   // 8x1024x1024 f32     (32 MB)
    _Float16* hA    = (_Float16*)(w + 46137344);   // 163840x64 f16       (20 MB)
    _Float16* hB    = hA + (size_t)163840 * 64;    // 163840x64 f16       (20 MB)
    _Float16* h2    = hA;                          // 163840x128 f16 overlays hA+hB

    // ---- Stage 1: kNN on 256-d features ----
    cvt_sq_kernel<<<32, 256, 0, stream>>>(feat, xh, sqn, 8192, 256);
    gram_dist_kernel<<<4096, 256, 0, stream>>>(xh, 256, 256, sqn, D, 64);
    topk_kernel<<<32, 256, 0, stream>>>(D, idx);

    // ---- EdgeConv1 layer 0: edge(512) -> 64 ----
    zero_kernel<<<1, 256, 0, stream>>>(gsum, 256);
    prep_w_plain<<<128, 256, 0, stream>>>(c1_w0, c1_b0, Wt, bEff, 512, 64);
    gemm_kernel<1><<<5120, 256, 65536, stream>>>(xh, idx, 256, 256, Wt, bEff, hA,
                                                 gsum, gsumsq, 512, 64, 4, 40960);
    bn_finalize<<<1, 64, 0, stream>>>(gsum, gsumsq, c1_g0, c1_be0, sArr, tArr, 64, 1.f / 163840.f);

    // ---- EdgeConv1 layer 1: 64 -> 64 (BN of layer0 folded into W) ----
    zero_kernel<<<1, 256, 0, stream>>>(gsum, 256);
    prep_w_folded<<<16, 256, 0, stream>>>(c1_w1, Wt, sArr, 64, 64);
    prep_b_folded<<<1, 64, 0, stream>>>(c1_w1, c1_b1, tArr, bEff, 64, 64);
    gemm_kernel<0><<<5120, 256, 8192, stream>>>(hA, nullptr, 0, 0, Wt, bEff, hB,
                                                gsum, gsumsq, 64, 64, 4, 40960);
    bn_finalize<<<1, 64, 0, stream>>>(gsum, gsumsq, c1_g1, c1_be1, sArr, tArr, 64, 1.f / 163840.f);

    // ---- EdgeConv1 layer 2: 64 -> 64 ----
    zero_kernel<<<1, 256, 0, stream>>>(gsum, 256);
    prep_w_folded<<<16, 256, 0, stream>>>(c1_w2, Wt, sArr, 64, 64);
    prep_b_folded<<<1, 64, 0, stream>>>(c1_w2, c1_b2, tArr, bEff, 64, 64);
    gemm_kernel<0><<<5120, 256, 8192, stream>>>(hB, nullptr, 0, 0, Wt, bEff, hA,
                                                gsum, gsumsq, 64, 64, 4, 40960);
    bn_finalize<<<1, 64, 0, stream>>>(gsum, gsumsq, c1_g2, c1_be2, sArr, tArr, 64, 1.f / 163840.f);

    // ---- Max aggregation -> x1 (hcat cols 0..63) ----
    aggregate_kernel<<<2048, 256, 0, stream>>>(hA, 64, sArr, tArr, hcat, 0, 8192);

    // ---- Stage 2: kNN on 64-d x1 ----
    sqnorm_h_kernel<<<32, 256, 0, stream>>>(hcat, 192, 64, sqn, 8192);
    gram_dist_kernel<<<4096, 256, 0, stream>>>(hcat, 192, 64, sqn, D, 64);
    topk_kernel<<<32, 256, 0, stream>>>(D, idx);

    // ---- EdgeConv2: edge(128) -> 128 ----
    zero_kernel<<<1, 256, 0, stream>>>(gsum, 256);
    prep_w_plain<<<64, 256, 0, stream>>>(c2_w0, c2_b0, Wt, bEff, 128, 128);
    gemm_kernel<1><<<10240, 256, 32768, stream>>>(hcat, idx, 192, 64, Wt, bEff, h2,
                                                  gsum, gsumsq, 128, 128, 8, 81920);
    bn_finalize<<<1, 128, 0, stream>>>(gsum, gsumsq, c2_g0, c2_be0, sArr, tArr, 128, 1.f / 163840.f);

    // ---- Max aggregation -> x2 (hcat cols 64..191) ----
    aggregate_kernel<<<4096, 256, 0, stream>>>(h2, 128, sArr, tArr, hcat, 64, 8192);

    // ---- Final layer: 192 -> 128 ----
    zero_kernel<<<1, 256, 0, stream>>>(gsum, 256);
    prep_w_plain<<<96, 256, 0, stream>>>(l_w, l_b, Wt, bEff, 192, 128);
    gemm_kernel<0><<<512, 256, 49152, stream>>>(hcat, nullptr, 0, 0, Wt, bEff, hF,
                                                gsum, gsumsq, 192, 128, 8, 4096);
    bn_finalize<<<1, 128, 0, stream>>>(gsum, gsumsq, l_g, l_be, sArr, tArr, 128, 1.f / 8192.f);
    apply_kernel<<<4096, 256, 0, stream>>>(hF, sArr, tArr, out, 8192 * 128, 128);
}
